// SparseSelfAttention_86861418594647
// MI455X (gfx1250) — matrix operand order repaired and checked by reading, not compiled, not run on hardware
//
#include <hip/hip_runtime.h>
#include <hip/hip_bf16.h>
#include <math.h>

// ---------------------------------------------------------------------------
// Sparse self-attention forward for MI455X (gfx1250), wave32 + WMMA bf16.
// B=2, T=1024, C=1024, H=16, D=64.  All GEMMs via v_wmma_f32_16x16x32_bf16.
// Vectorized (b128) staging, async global->LDS copies where layouts match.
// ---------------------------------------------------------------------------

typedef __bf16 bf16;
typedef __attribute__((ext_vector_type(16))) __bf16 v16bf;
typedef __attribute__((ext_vector_type(8)))  __bf16 v8bf;
typedef __attribute__((ext_vector_type(4)))  __bf16 v4bf;
typedef __attribute__((ext_vector_type(8)))  float  v8f;
typedef __attribute__((ext_vector_type(4)))  float  v4f;

union Frag16 {            // 8 VGPRs viewed as one WMMA 16-bit operand
    v16bf v;
    v8bf  h[2];
};

constexpr int Bb = 2;
constexpr int Tt = 1024;
constexpr int Cc = 1024;
constexpr int Hh = 16;
constexpr int Dd = 64;       // head dim
constexpr int N3 = 3 * Cc;   // 3072

constexpr int TILE_M = 128;
constexpr int TILE_N = 64;
constexpr int TILE_K = 32;

// A fragment from [row][K] layout (one 32B contiguous run per lane)
__device__ __forceinline__ Frag16 load_a_frag(const bf16* rowp, int hi) {
    Frag16 f;
    f.v = *(const v16bf*)(rowp + hi * 16);         // K = 16*hi .. +15
    return f;
}
// B fragment from [col][K] layout (one 32B contiguous run per lane)
__device__ __forceinline__ v16bf load_b_frag(const bf16* colp, int hi) {
    return *(const v16bf*)(colp + hi * 16);        // K = 16*hi .. +15
}

// CDNA5 async global->LDS copy, 16B per lane (ISA §15.18.3 op 98).
// vdst = LDS byte address (generic ptr low 32 bits per aperture rule).
__device__ __forceinline__ void async_ld_b128(void* lds, const void* gptr) {
    asm volatile("global_load_async_to_lds_b128 %0, %1, off"
                 :: "v"((uint32_t)(uintptr_t)lds),
                    "v"((uint64_t)(uintptr_t)gptr)
                 : "memory");
}
__device__ __forceinline__ void wait_async0() {
    asm volatile("s_wait_asynccnt 0x0" ::: "memory");
}

// ---------------------------------------------------------------------------
// Kernel 1: qkv = x @ W_attn + b_attn  -> q/k/v head-major bf16 in workspace
// ---------------------------------------------------------------------------
__global__ __launch_bounds__(256) void qkv_gemm_kernel(
    const float* __restrict__ x, const float* __restrict__ W,
    const float* __restrict__ bias,
    bf16* __restrict__ qh, bf16* __restrict__ kh, bf16* __restrict__ vh)
{
    __shared__ bf16 As [TILE_M][TILE_K];   // [row][K]  8 KB
    __shared__ bf16 BsT[TILE_N][TILE_K];   // [col][K]  4 KB (transposed)

    const int tid  = threadIdx.x;
    const int lane = tid & 31;
    const int wave = tid >> 5;
    const int ln15 = lane & 15;
    const int hi   = lane >> 4;

    const int nblocks = N3 / TILE_N;      // 48
    const int m0 = (blockIdx.x / nblocks) * TILE_M;
    const int n0 = (blockIdx.x % nblocks) * TILE_N;
    const int wm = (wave & 3) * 32;
    const int wn = (wave >> 2) * 32;

    v8f acc[2][2] = {};

    for (int kk = 0; kk < Cc; kk += TILE_K) {
        // A tile: 4096 f32->bf16, 8 per thread, 2 passes, b128 in / b128 out
        #pragma unroll
        for (int it = 0; it < 2; ++it) {
            int i = it * 2048 + tid * 8;
            int r = i >> 5, c = i & 31;                    // c in {0,8,16,24}
            const float* src = &x[(size_t)(m0 + r) * Cc + kk + c];
            v4f lo = *(const v4f*)(src);
            v4f hi4 = *(const v4f*)(src + 4);
            v8bf o;
            #pragma unroll
            for (int j = 0; j < 4; ++j) { o[j] = (bf16)lo[j]; o[j + 4] = (bf16)hi4[j]; }
            *(v8bf*)&As[r][c] = o;
        }
        // B tile: 2048 f32->bf16 transposed, 4 per thread, 2 passes
        #pragma unroll
        for (int it = 0; it < 2; ++it) {
            int i = it * 1024 + tid * 4;
            int r = i >> 6, c = i & 63;                    // c multiple of 4
            v4f wv = *(const v4f*)&W[(size_t)(kk + r) * N3 + n0 + c];
            #pragma unroll
            for (int j = 0; j < 4; ++j) BsT[c + j][r] = (bf16)wv[j];
        }
        __syncthreads();

        if (kk + TILE_K < Cc) {
            __builtin_prefetch(&x[(size_t)(m0 + (tid >> 1)) * Cc + kk + TILE_K], 0, 1);
            __builtin_prefetch(&W[(size_t)(kk + TILE_K + (tid >> 3)) * N3 + n0], 0, 1);
        }

        #pragma unroll
        for (int sm = 0; sm < 2; ++sm) {
            Frag16 a = load_a_frag(&As[wm + sm * 16 + ln15][0], hi);
            #pragma unroll
            for (int sn = 0; sn < 2; ++sn) {
                v16bf b = load_b_frag(&BsT[wn + sn * 16 + ln15][0], hi);
                acc[sm][sn] = __builtin_amdgcn_wmma_f32_16x16x32_bf16(
                    false, a.v, false, b, (short)0, acc[sm][sn], false, false);
            }
        }
        __syncthreads();
    }

    // Epilogue.  TILE_N==64==D: the whole block maps to one (sel, head).
    const int sel = n0 >> 10;                 // 0=q 1=k 2=v
    const int h   = (n0 & (Cc - 1)) >> 6;     // head
    const int b   = m0 >> 10;                 // batch (TILE_M | T)
    const int t0  = m0 & (Tt - 1);
    bf16* __restrict__ dstp = (sel == 0) ? qh : (sel == 1) ? kh : vh;
    dstp += ((size_t)b * Hh + h) * Tt * Dd;

    #pragma unroll
    for (int sm = 0; sm < 2; ++sm) {
        #pragma unroll
        for (int sn = 0; sn < 2; ++sn) {
            int d = wn + sn * 16 + ln15;                       // 0..63
            float bb = bias[n0 + d];
            #pragma unroll
            for (int i = 0; i < 8; ++i) {
                int t = t0 + wm + sm * 16 + i + (hi ? 8 : 0);
                dstp[(size_t)t * Dd + d] = (bf16)(acc[sm][sn][i] + bb);
            }
        }
    }
}

// ---------------------------------------------------------------------------
// Kernel 2: attention.  One wave owns 16 query rows of one (b,h).
// ---------------------------------------------------------------------------
__global__ __launch_bounds__(256) void attn_kernel(
    const bf16* __restrict__ qh, const bf16* __restrict__ kh,
    const bf16* __restrict__ vh, const float* __restrict__ noise,
    const int* __restrict__ is_causal_p, const int* __restrict__ sample_p,
    float* __restrict__ att_out, bf16* __restrict__ ymerged)
{
    __shared__ bf16  wtile[8][16][32];  // per-wave softmax*edge, [row][K], 8 KB
    __shared__ bf16  Vt[Dd][32];        // V^T for current K block, [col][K], 4 KB
    __shared__ float mrow[8][16];       // per-wave row max
    __shared__ float srow[8][16];       // per-wave row sum

    const int tid  = threadIdx.x;
    const int lane = tid & 31;
    const int wave = tid >> 5;
    const int ln15 = lane & 15;
    const int hi   = lane >> 4;

    const int qtiles = Tt / 128;
    const int bh = blockIdx.x / qtiles;
    const int qt = blockIdx.x % qtiles;
    const int b  = bh / Hh;
    const int h  = bh % Hh;
    const int q0 = qt * 128 + wave * 16;

    const int causal = is_causal_p[0];
    const int sample = sample_p[0];

    const size_t basehd   = (size_t)bh * Tt * Dd;
    const size_t att_base = (size_t)bh * Tt * Tt;
    const float  scale    = 0.125f;              // 1/sqrt(64)
    const float  NEG_BIG  = -3.0e38f;            // finite -inf stand-in

    // Q fragments over head dim: d in [0,32) and [32,64)
    const bf16* qrow = qh + basehd + (size_t)(q0 + ln15) * Dd;
    Frag16 qf0 = load_a_frag(qrow, hi);
    Frag16 qf1 = load_a_frag(qrow + 32, hi);

    float m[8], s[8];
    #pragma unroll
    for (int i = 0; i < 8; ++i) { m[i] = NEG_BIG; s[i] = 0.0f; }

    // ---- Pass 1: scores + mask + att write + online (max,sum) -------------
    for (int kt = 0; kt < Tt / 16; ++kt) {
        const bf16* krow = kh + basehd + (size_t)(kt * 16 + ln15) * Dd;
        v16bf kb0 = *(const v16bf*)(krow + hi * 16);
        v16bf kb1 = *(const v16bf*)(krow + 32 + hi * 16);

        v8f c = {};
        c = __builtin_amdgcn_wmma_f32_16x16x32_bf16(false, qf0.v, false, kb0,
                                                    (short)0, c, false, false);
        c = __builtin_amdgcn_wmma_f32_16x16x32_bf16(false, qf1.v, false, kb1,
                                                    (short)0, c, false, false);
        #pragma unroll
        for (int i = 0; i < 8; ++i) {
            int r   = q0 + i + (hi ? 8 : 0);
            int col = kt * 16 + ln15;
            float v = c[i] * scale;
            bool masked = causal && (col > r);
            att_out[att_base + (size_t)r * Tt + col] = masked ? -INFINITY : v;
            float vi = masked ? NEG_BIG : v;
            float nm = fmaxf(m[i], vi);
            s[i] = s[i] * __expf(m[i] - nm) + __expf(vi - nm);
            m[i] = nm;
        }
    }
    // combine (m,s) across the 16 lanes sharing each row
    #pragma unroll
    for (int i = 0; i < 8; ++i) {
        #pragma unroll
        for (int off = 8; off >= 1; off >>= 1) {
            float mo = __shfl_xor(m[i], off, 16);
            float so = __shfl_xor(s[i], off, 16);
            float nm = fmaxf(m[i], mo);
            s[i] = s[i] * __expf(m[i] - nm) + so * __expf(mo - nm);
            m[i] = nm;
        }
    }
    // publish row stats (identical across the 16 lanes of each half-group)
    #pragma unroll
    for (int i = 0; i < 8; ++i) {
        mrow[wave][i + hi * 8] = m[i];
        srow[wave][i + hi * 8] = s[i];
    }
    __syncthreads();

    // This lane's row (ln15) and its 16-column slab (hi) for w computation
    const float myM = mrow[wave][ln15];
    const float myS = srow[wave][ln15];

    // ---- Pass 2: w = softmax * edge;  y = w @ V ---------------------------
    v8f accY[4] = {};
    for (int kk = 0; kk < Tt; kk += 32) {
        // Stage V^T for this K block (shared across waves, coalesced b128 in)
        {
            int r  = tid >> 3;            // 0..31 key row
            int d0 = (tid & 7) * 8;       // 0..56 head-dim start
            v8bf vv = *(const v8bf*)(vh + basehd + (size_t)(kk + r) * Dd + d0);
            #pragma unroll
            for (int j = 0; j < 8; ++j) Vt[d0 + j][r] = vv[j];
        }
        // w-tile: lane handles row ln15, 16 contiguous columns (64B loads)
        {
            const size_t rowoff = att_base + (size_t)(q0 + ln15) * Tt + kk + hi * 16;
            const float* arow = att_out + rowoff;
            const float* nrow = noise + rowoff;
            #pragma unroll
            for (int g = 0; g < 4; ++g) {
                v4f av = *(const v4f*)(arow + g * 4);
                v4f uv = {};
                if (sample) uv = *(const v4f*)(nrow + g * 4);
                v4bf wo;
                #pragma unroll
                for (int j = 0; j < 4; ++j) {
                    float a = av[j];
                    float p = __expf(a - myM) / myS;
                    float edge;
                    if (sample) {
                        float u  = uv[j];
                        float lg = __logf(u) - log1pf(-u);     // logit(u)
                        edge = (a + lg > 0.0f) ? 1.0f : 0.0f;  // straight-through
                    } else {
                        edge = (a > 0.0f) ? 1.0f : 0.0f;
                    }
                    wo[j] = (bf16)(p * edge);
                }
                *(v4bf*)&wtile[wave][ln15][hi * 16 + g * 4] = wo;
            }
        }
        __syncthreads();

        Frag16 aw = load_a_frag(&wtile[wave][ln15][0], hi);
        #pragma unroll
        for (int nd = 0; nd < 4; ++nd) {
            v16bf bv = load_b_frag(&Vt[nd * 16 + ln15][0], hi);
            accY[nd] = __builtin_amdgcn_wmma_f32_16x16x32_bf16(
                false, aw.v, false, bv, (short)0, accY[nd], false, false);
        }
        __syncthreads();
    }

    #pragma unroll
    for (int nd = 0; nd < 4; ++nd) {
        #pragma unroll
        for (int i = 0; i < 8; ++i) {
            int t    = q0 + i + (hi ? 8 : 0);
            int cdim = h * Dd + nd * 16 + ln15;
            ymerged[((size_t)b * Tt + t) * Cc + cdim] = (bf16)accY[nd][i];
        }
    }
}

// ---------------------------------------------------------------------------
// Kernel 3: y = ymerged @ W_proj + b_proj  (f32 out).
// A tile staged with CDNA5 async global->LDS (byte-exact bf16 copy).
// ---------------------------------------------------------------------------
__global__ __launch_bounds__(256) void proj_gemm_kernel(
    const bf16* __restrict__ A, const float* __restrict__ W,
    const float* __restrict__ bias, float* __restrict__ out)
{
    __shared__ bf16 As [TILE_M][TILE_K];
    __shared__ bf16 BsT[TILE_N][TILE_K];

    const int tid  = threadIdx.x;
    const int lane = tid & 31;
    const int wave = tid >> 5;
    const int ln15 = lane & 15;
    const int hi   = lane >> 4;

    const int nblocks = Cc / TILE_N;      // 16
    const int m0 = (blockIdx.x / nblocks) * TILE_M;
    const int n0 = (blockIdx.x % nblocks) * TILE_N;
    const int wm = (wave & 3) * 32;
    const int wn = (wave >> 2) * 32;

    v8f acc[2][2] = {};

    for (int kk = 0; kk < Cc; kk += TILE_K) {
        // A tile: async DMA, 16B per thread-slot, 2 passes (8 KB total)
        #pragma unroll
        for (int it = 0; it < 2; ++it) {
            int i = it * 2048 + tid * 8;
            int r = i >> 5, c = i & 31;
            async_ld_b128(&As[r][c], &A[(size_t)(m0 + r) * Cc + kk + c]);
        }
        // B tile: f32->bf16 transposed
        #pragma unroll
        for (int it = 0; it < 2; ++it) {
            int i = it * 1024 + tid * 4;
            int r = i >> 6, c = i & 63;
            v4f wv = *(const v4f*)&W[(size_t)(kk + r) * Cc + n0 + c];
            #pragma unroll
            for (int j = 0; j < 4; ++j) BsT[c + j][r] = (bf16)wv[j];
        }
        wait_async0();
        __syncthreads();

        if (kk + TILE_K < Cc) {
            __builtin_prefetch(&W[(size_t)(kk + TILE_K + (tid >> 3)) * Cc + n0], 0, 1);
        }

        #pragma unroll
        for (int sm = 0; sm < 2; ++sm) {
            Frag16 a = load_a_frag(&As[wm + sm * 16 + ln15][0], hi);
            #pragma unroll
            for (int sn = 0; sn < 2; ++sn) {
                v16bf b = load_b_frag(&BsT[wn + sn * 16 + ln15][0], hi);
                acc[sm][sn] = __builtin_amdgcn_wmma_f32_16x16x32_bf16(
                    false, a.v, false, b, (short)0, acc[sm][sn], false, false);
            }
        }
        __syncthreads();
    }

    #pragma unroll
    for (int sm = 0; sm < 2; ++sm) {
        #pragma unroll
        for (int sn = 0; sn < 2; ++sn) {
            int n = n0 + wn + sn * 16 + ln15;
            float bb = bias[n];
            #pragma unroll
            for (int i = 0; i < 8; ++i) {
                int r = m0 + wm + sm * 16 + i + (hi ? 8 : 0);
                out[(size_t)r * Cc + n] = acc[sm][sn][i] + bb;
            }
        }
    }
}

// ---------------------------------------------------------------------------
// Host launcher
// ---------------------------------------------------------------------------
extern "C" void kernel_launch(void* const* d_in, const int* in_sizes, int n_in,
                              void* d_out, int out_size, void* d_ws, size_t ws_size,
                              hipStream_t stream)
{
    const float* x      = (const float*)d_in[0];
    const float* W_attn = (const float*)d_in[1];
    const float* b_attn = (const float*)d_in[2];
    const float* W_proj = (const float*)d_in[3];
    const float* b_proj = (const float*)d_in[4];
    const float* noise  = (const float*)d_in[5];
    const int*   is_causal = (const int*)d_in[6];
    const int*   sample    = (const int*)d_in[7];

    float* y_out   = (float*)d_out;                       // [B,T,C]
    float* att_out = y_out + (size_t)Bb * Tt * Cc;        // [B,H,T,T]

    const size_t headElems = (size_t)Bb * Hh * Tt * Dd;
    bf16* qh = (bf16*)d_ws;
    bf16* kh = qh + headElems;
    bf16* vh = kh + headElems;
    bf16* ym = vh + headElems;

    qkv_gemm_kernel<<<dim3((N3 / TILE_N) * ((Bb * Tt) / TILE_M)), dim3(256), 0, stream>>>(
        x, W_attn, b_attn, qh, kh, vh);

    attn_kernel<<<dim3(Bb * Hh * (Tt / 128)), dim3(256), 0, stream>>>(
        qh, kh, vh, noise, is_causal, sample, att_out, ym);

    proj_gemm_kernel<<<dim3((Cc / TILE_N) * ((Bb * Tt) / TILE_M)), dim3(256), 0, stream>>>(
        ym, W_proj, b_proj, y_out);
}
